// AtomToTokenCrossAttn_27693949125123
// MI455X (gfx1250) — compile-verified, matrix operand order checked by reading, not run on hardware
//
#include <hip/hip_runtime.h>
#include <hip/hip_bf16.h>
#include <stdint.h>

// ---------------------------------------------------------------------------
// AtomToTokenCrossAttn for MI455X (gfx1250, wave32).
// GEMMs on v_wmma_f32_16x16x32_bf16. A-tiles staged to LDS with
// global_load_async_to_lds_b128 (ASYNCcnt), double-buffered in the Q/G GEMM.
// Weight (B) fragments held in registers across all row tiles of a block.
// BF16 C tiles repacked through a per-wave LDS stash -> coalesced b128 stores.
// Ragged windowed attention on VALU with online softmax.
// ---------------------------------------------------------------------------

typedef __bf16 bf16_t;
typedef __attribute__((ext_vector_type(16))) __bf16 v16bf;
typedef __attribute__((ext_vector_type(8)))  __bf16 v8bf;   // 16 bytes
typedef __attribute__((ext_vector_type(8)))  float  v8f;

// Problem dims (match reference)
constexpr int Bb     = 4;
constexpr int Nn     = 4096;
constexpr int Mm     = 32768;
constexpr int D_TOK  = 512;
constexpr int D_ATOM = 128;
constexpr int NT     = Bb * Nn;   // 16384 tokens
constexpr int NA     = Bb * Mm;   // 131072 atoms
constexpr float LN_EPS = 1e-5f;

// ---------------------------------------------------------------------------
// helpers
// ---------------------------------------------------------------------------
__device__ inline uint32_t lds_off(const void* p) {
  // generic LDS pointer: low 32 bits are the LDS byte offset (ISA 10.2)
  return (uint32_t)(uintptr_t)p;
}

// Async-copy a [rows x row_bytes] sub-block (row stride ld_elems bf16) from
// global to LDS laid out densely row-major. Per-lane 16B transfers on ASYNCcnt.
__device__ inline void async_stage_2d(const bf16_t* __restrict__ g, int ld_elems,
                                      int rows, int row_bytes, uint32_t lds_base,
                                      int tid, int nth) {
  int cpr    = row_bytes >> 4;   // 16B chunks per row
  int chunks = rows * cpr;
  for (int c = tid; c < chunks; c += nth) {
    int r = c / cpr, k = c - r * cpr;
    uint32_t loff = lds_base + (uint32_t)(c << 4);
    const void* gp = (const char*)(g + (size_t)r * ld_elems) + (k << 4);
    asm volatile("global_load_async_to_lds_b128 %0, %1, off"
                 :: "v"(loff), "v"(gp) : "memory");
  }
}
__device__ inline void async_wait_all() {
  asm volatile("s_wait_asynccnt 0" ::: "memory");
}

// A fragment (16x32 bf16) from LDS tile laid out row-major [16][K].
// Contiguous k order chosen here (same as load_b_frag, not the ISA table order): lanes 0-15 hold row M=lane, K=k0..k0+15; lanes 16-31 hold
// row M=lane-16, K=k0+16..k0+31. 16 contiguous halves per lane.
__device__ inline v16bf load_a_frag_lds(const bf16_t* tile, int K, int k0, int lane) {
  int row = lane & 15;
  int kk  = k0 + ((lane >> 4) << 4);
  return *(const v16bf*)(tile + row * K + kk);
}

// B fragment (32x16 bf16) from a column-major (pre-transposed) weight
// Wt[Ncols][K]: lane owns output column n0+(lane&15); halves contiguous.
__device__ inline v16bf load_b_frag(const bf16_t* __restrict__ wt, int K,
                                    int n0, int k0, int lane) {
  int col = n0 + (lane & 15);
  int kk  = k0 + ((lane >> 4) << 4);
  return *(const v16bf*)(wt + (size_t)col * K + kk);
}

__device__ inline v8f bwmma(v16bf a, v16bf b, v8f c) {
  return __builtin_amdgcn_wmma_f32_16x16x32_bf16(
      /*neg_a=*/false, a, /*neg_b=*/false, b,
      /*c_mod=*/(short)0, c, /*reuse_a=*/false, /*reuse_b=*/false);
}

// C/D layout (ISA 7.12.2): VGPR r, lanes 0-15 -> (M=r, N=lane),
// lanes 16-31 -> (M=8+r, N=lane-16).
__device__ inline void store_c_f32(float* __restrict__ out, int ldc,
                                   int m0, int n0, int lane, v8f c) {
  int col   = n0 + (lane & 15);
  int mbase = m0 + ((lane >> 4) << 3);
#pragma unroll
  for (int r = 0; r < 8; ++r)
    out[(size_t)(mbase + r) * ldc + col] = c[r];
}

// BF16 C store: repack the 16x16 tile through a per-wave LDS stash
// (row-major [16][16] bf16), then one coalesced 16B global store per lane
// (wave covers the 512B tile in a single b128 store instruction).
// LDS store->load from the same wave is in-order on DScnt (ISA 7.3), and the
// stash is wave-private, so no barrier is required.
__device__ inline void store_c_bf16_coal(bf16_t* __restrict__ out, int ldc,
                                         int m0, int n0, int lane, v8f c,
                                         bf16_t* __restrict__ stash) {
  int col   = lane & 15;
  int rbase = (lane >> 4) << 3;
#pragma unroll
  for (int r = 0; r < 8; ++r)
    stash[(rbase + r) * 16 + col] = (bf16_t)c[r];
  int row  = lane >> 1;          // 2 lanes per row
  int half = (lane & 1) << 3;    // 8-element halves
  v8bf rv = *(const v8bf*)(stash + row * 16 + half);
  *(v8bf*)(out + (size_t)(m0 + row) * ldc + n0 + half) = rv;
}

// ---------------------------------------------------------------------------
// weight transpose + fp32 -> bf16  (tiny)
// wt[c * rows + r] = w[r * cols + c]   -> column-major bf16 for B fragments
// ---------------------------------------------------------------------------
__global__ void transpose_to_bf16(const float* __restrict__ w,
                                  bf16_t* __restrict__ wt, int rows, int cols) {
  int idx = blockIdx.x * blockDim.x + threadIdx.x;
  if (idx >= rows * cols) return;
  int r = idx / cols, c = idx % cols;
  wt[(size_t)c * rows + r] = (bf16_t)w[idx];
}

// ---------------------------------------------------------------------------
// LayerNorm (wave per row) -> bf16
// ---------------------------------------------------------------------------
template <int D>
__global__ __launch_bounds__(256) void ln_to_bf16(const float* __restrict__ x,
                                                  const float* __restrict__ g,
                                                  const float* __restrict__ b,
                                                  bf16_t* __restrict__ y, int rows) {
  int wave = (int)((blockIdx.x * blockDim.x + threadIdx.x) >> 5);
  int lane = threadIdx.x & 31;
  if (wave >= rows) return;
  const float* xr = x + (size_t)wave * D;
  constexpr int PER = D / 32;
  float v[PER];
  float s = 0.f, s2 = 0.f;
#pragma unroll
  for (int i = 0; i < PER; ++i) {
    v[i] = xr[lane + i * 32];
    s += v[i];
    s2 += v[i] * v[i];
  }
#pragma unroll
  for (int off = 16; off > 0; off >>= 1) {
    s  += __shfl_xor(s, off, 32);
    s2 += __shfl_xor(s2, off, 32);
  }
  float mu  = s / (float)D;
  float var = s2 / (float)D - mu * mu;
  float rs  = rsqrtf(var + LN_EPS);
  bf16_t* yr = y + (size_t)wave * D;
#pragma unroll
  for (int i = 0; i < PER; ++i) {
    int d = lane + i * 32;
    yr[d] = (bf16_t)((v[i] - mu) * rs * g[d] + b[d]);
  }
}

// ---------------------------------------------------------------------------
// K/V GEMM: (NA x 128) x (128 x 128) twice.
// Block = 128 atoms (8 M-tiles); 8 waves each own one 16-col slice.
// B-fragments for w_k and w_v (4 k-steps each) held in registers for the
// whole block -> one weight fetch per 8 M-tiles. A streamed via LDS.
// ---------------------------------------------------------------------------
__global__ __launch_bounds__(256) void kv_gemm(const bf16_t* __restrict__ an,
                                               const bf16_t* __restrict__ wkt,
                                               const bf16_t* __restrict__ wvt,
                                               bf16_t* __restrict__ Kout,
                                               bf16_t* __restrict__ Vout) {
  __shared__ __align__(32) bf16_t tileA[128 * 128];
  __shared__ __align__(32) bf16_t stash[8][16 * 16];   // per-wave C-tile bounce
  int m0  = blockIdx.x * 128;
  int tid = threadIdx.x;
  async_stage_2d(an + (size_t)m0 * 128, 128, 128, 256, lds_off(tileA), tid, 256);

  int wave = tid >> 5, lane = tid & 31;
  int n0 = wave * 16;
  v16bf bk[4], bv[4];
#pragma unroll
  for (int ks = 0; ks < 4; ++ks) {
    bk[ks] = load_b_frag(wkt, 128, n0, ks * 32, lane);
    bv[ks] = load_b_frag(wvt, 128, n0, ks * 32, lane);
  }
  async_wait_all();
  __syncthreads();

#pragma unroll
  for (int mt = 0; mt < 8; ++mt) {
    v8f ck = {}, cv = {};
#pragma unroll
    for (int ks = 0; ks < 4; ++ks) {
      v16bf af = load_a_frag_lds(tileA + mt * 16 * 128, 128, ks * 32, lane);
      ck = bwmma(af, bk[ks], ck);
      cv = bwmma(af, bv[ks], cv);
    }
    store_c_bf16_coal(Kout, 128, m0 + mt * 16, n0, lane, ck, stash[wave]);
    store_c_bf16_coal(Vout, 128, m0 + mt * 16, n0, lane, cv, stash[wave]);
  }
}

// ---------------------------------------------------------------------------
// Q/G GEMM: (NT x 512) x (512 x 128) twice, FP32 outputs.
// Block = 64 tokens (4 M-tiles) x 128 cols. K processed in 4 chunks of 128
// with DOUBLE-BUFFERED async LDS staging: chunk i+1 copies while chunk i
// feeds WMMAs. Accumulators persist across K (4 mt x 2 outs x v8f).
// ---------------------------------------------------------------------------
__global__ __launch_bounds__(256) void qg_gemm(const bf16_t* __restrict__ sn,
                                               const bf16_t* __restrict__ wqt,
                                               const bf16_t* __restrict__ wgt,
                                               float* __restrict__ Qf,
                                               float* __restrict__ Gf) {
  __shared__ __align__(32) bf16_t tileA[2][64 * 128];
  int m0  = blockIdx.x * 64;
  int tid = threadIdx.x;
  int wave = tid >> 5, lane = tid & 31;
  int n0 = wave * 16;

  const bf16_t* abase = sn + (size_t)m0 * 512;
  // prologue: stage chunk 0
  async_stage_2d(abase, 512, 64, 256, lds_off(tileA[0]), tid, 256);
  async_wait_all();
  __syncthreads();

  v8f cq[4] = {}, cg[4] = {};
#pragma unroll
  for (int kc = 0; kc < 4; ++kc) {
    const bf16_t* cur = tileA[kc & 1];
    // prefetch next K-chunk while computing on current one
    if (kc < 3)
      async_stage_2d(abase + (kc + 1) * 128, 512, 64, 256,
                     lds_off(tileA[(kc + 1) & 1]), tid, 256);
#pragma unroll
    for (int ks = 0; ks < 4; ++ks) {
      int kglob = kc * 128 + ks * 32;
      v16bf bq = load_b_frag(wqt, 512, n0, kglob, lane);
      v16bf bg = load_b_frag(wgt, 512, n0, kglob, lane);
#pragma unroll
      for (int mt = 0; mt < 4; ++mt) {
        v16bf af = load_a_frag_lds(cur + mt * 16 * 128, 128, ks * 32, lane);
        cq[mt] = bwmma(af, bq, cq[mt]);
        cg[mt] = bwmma(af, bg, cg[mt]);
      }
    }
    async_wait_all();
    __syncthreads();
  }
#pragma unroll
  for (int mt = 0; mt < 4; ++mt) {
    store_c_f32(Qf, 128, m0 + mt * 16, n0, lane, cq[mt]);
    store_c_f32(Gf, 128, m0 + mt * 16, n0, lane, cg[mt]);
  }
}

// ---------------------------------------------------------------------------
// Ragged windowed attention + mask + sigmoid gate. One wave per token.
// Lane l owns dim l of each of the 4 heads; online softmax over <=16 atoms.
// ---------------------------------------------------------------------------
__global__ __launch_bounds__(256) void attn_kernel(
    const float* __restrict__ Qf, const float* __restrict__ Gf,
    const bf16_t* __restrict__ Kb, const bf16_t* __restrict__ Vb,
    const int* __restrict__ starts, const int* __restrict__ counts,
    const float* __restrict__ tmask, bf16_t* __restrict__ attg) {
  int tok = (int)((blockIdx.x * blockDim.x + threadIdx.x) >> 5);
  int lane = threadIdx.x & 31;
  if (tok >= NT) return;
  int b     = tok / Nn;
  int start = starts[tok];
  int cnt   = counts[tok];
  float tm  = tmask[tok];
  const float scale = 0.17677669529663687f;  // 1/sqrt(32)

  float q[4];
#pragma unroll
  for (int h = 0; h < 4; ++h) q[h] = Qf[(size_t)tok * 128 + h * 32 + lane];

  float m[4], l[4], acc[4];
#pragma unroll
  for (int h = 0; h < 4; ++h) {
    m[h] = -__builtin_huge_valf(); l[h] = 0.f; acc[h] = 0.f;
  }

  for (int w = 0; w < cnt; ++w) {
    size_t row = (size_t)b * Mm + (size_t)(start + w);
    const bf16_t* kr = Kb + row * 128;
    const bf16_t* vr = Vb + row * 128;
    if (w + 1 < cnt) __builtin_prefetch(kr + 128, 0, 0);  // global_prefetch_b8
    float sh[4], vv[4];
#pragma unroll
    for (int h = 0; h < 4; ++h) {
      sh[h] = q[h] * (float)kr[h * 32 + lane];
      vv[h] = (float)vr[h * 32 + lane];
    }
#pragma unroll
    for (int off = 16; off > 0; off >>= 1)
#pragma unroll
      for (int h = 0; h < 4; ++h) sh[h] += __shfl_xor(sh[h], off, 32);
#pragma unroll
    for (int h = 0; h < 4; ++h) {
      float s    = sh[h] * scale;
      float mn   = fmaxf(m[h], s);
      float corr = __expf(m[h] - mn);
      float p    = __expf(s - mn);
      l[h]   = l[h] * corr + p;
      acc[h] = acc[h] * corr + p * vv[h];
      m[h]   = mn;
    }
  }
  bf16_t* outr = attg + (size_t)tok * 128;
#pragma unroll
  for (int h = 0; h < 4; ++h) {
    float att  = (l[h] > 0.f ? acc[h] / l[h] : 0.f) * tm;
    float g    = Gf[(size_t)tok * 128 + h * 32 + lane];
    float gate = 1.f / (1.f + __expf(-g));
    outr[h * 32 + lane] = (bf16_t)(gate * att);
  }
}

// ---------------------------------------------------------------------------
// Output GEMM: (NT x 128) x (128 x 512) -> FP32 out.
// Block = 64 tokens (4 M-tiles); grid.y = 128-col group. B-frags (4) in
// registers for the whole block.
// ---------------------------------------------------------------------------
__global__ __launch_bounds__(256) void out_gemm(const bf16_t* __restrict__ attg,
                                                const bf16_t* __restrict__ wot,
                                                float* __restrict__ out) {
  __shared__ __align__(32) bf16_t tileA[64 * 128];
  int m0  = blockIdx.x * 64;
  int tid = threadIdx.x;
  async_stage_2d(attg + (size_t)m0 * 128, 128, 64, 256, lds_off(tileA), tid, 256);

  int wave = tid >> 5, lane = tid & 31;
  int n0 = blockIdx.y * 128 + wave * 16;
  v16bf bo[4];
#pragma unroll
  for (int ks = 0; ks < 4; ++ks) bo[ks] = load_b_frag(wot, 128, n0, ks * 32, lane);
  async_wait_all();
  __syncthreads();

#pragma unroll
  for (int mt = 0; mt < 4; ++mt) {
    v8f c = {};
#pragma unroll
    for (int ks = 0; ks < 4; ++ks) {
      v16bf af = load_a_frag_lds(tileA + mt * 16 * 128, 128, ks * 32, lane);
      c = bwmma(af, bo[ks], c);
    }
    store_c_f32(out, 512, m0 + mt * 16, n0, lane, c);
  }
}

// ---------------------------------------------------------------------------
// launch
// ---------------------------------------------------------------------------
extern "C" void kernel_launch(void* const* d_in, const int* in_sizes, int n_in,
                              void* d_out, int out_size, void* d_ws, size_t ws_size,
                              hipStream_t stream) {
  const float* s      = (const float*)d_in[0];
  const float* a      = (const float*)d_in[1];
  const int*   starts = (const int*)d_in[2];
  const int*   counts = (const int*)d_in[3];
  const float* tmask  = (const float*)d_in[4];
  const float* w_q    = (const float*)d_in[5];
  const float* w_k    = (const float*)d_in[6];
  const float* w_v    = (const float*)d_in[7];
  const float* w_g    = (const float*)d_in[8];
  const float* w_o    = (const float*)d_in[9];
  const float* lnqg   = (const float*)d_in[10];
  const float* lnqb   = (const float*)d_in[11];
  const float* lnkg   = (const float*)d_in[12];
  const float* lnkb   = (const float*)d_in[13];
  float* out = (float*)d_out;

  char* ws = (char*)d_ws;
  size_t off = 0;
  auto alloc = [&](size_t bytes) {
    char* p = ws + off;
    off = (off + bytes + 255) & ~(size_t)255;
    return p;
  };
  bf16_t* sn   = (bf16_t*)alloc((size_t)NT * D_TOK * 2);   // 16 MB
  bf16_t* an   = (bf16_t*)alloc((size_t)NA * D_ATOM * 2);  // 32 MB
  bf16_t* Kb   = (bf16_t*)alloc((size_t)NA * D_ATOM * 2);  // 32 MB
  bf16_t* Vb   = (bf16_t*)alloc((size_t)NA * D_ATOM * 2);  // 32 MB
  float*  Qf   = (float*) alloc((size_t)NT * D_ATOM * 4);  // 8 MB
  float*  Gf   = (float*) alloc((size_t)NT * D_ATOM * 4);  // 8 MB
  bf16_t* attg = (bf16_t*)alloc((size_t)NT * D_ATOM * 2);  // 4 MB
  bf16_t* wqt  = (bf16_t*)alloc((size_t)D_TOK * D_ATOM * 2);
  bf16_t* wgt  = (bf16_t*)alloc((size_t)D_TOK * D_ATOM * 2);
  bf16_t* wkt  = (bf16_t*)alloc((size_t)D_ATOM * D_ATOM * 2);
  bf16_t* wvt  = (bf16_t*)alloc((size_t)D_ATOM * D_ATOM * 2);
  bf16_t* wot  = (bf16_t*)alloc((size_t)D_ATOM * D_TOK * 2);
  (void)in_sizes; (void)n_in; (void)out_size; (void)ws_size;

  // 1) weights -> bf16 column-major
  transpose_to_bf16<<<(D_TOK * D_ATOM + 255) / 256, 256, 0, stream>>>(w_q, wqt, D_TOK, D_ATOM);
  transpose_to_bf16<<<(D_TOK * D_ATOM + 255) / 256, 256, 0, stream>>>(w_g, wgt, D_TOK, D_ATOM);
  transpose_to_bf16<<<(D_ATOM * D_ATOM + 255) / 256, 256, 0, stream>>>(w_k, wkt, D_ATOM, D_ATOM);
  transpose_to_bf16<<<(D_ATOM * D_ATOM + 255) / 256, 256, 0, stream>>>(w_v, wvt, D_ATOM, D_ATOM);
  transpose_to_bf16<<<(D_ATOM * D_TOK + 255) / 256, 256, 0, stream>>>(w_o, wot, D_ATOM, D_TOK);

  // 2) LayerNorms -> bf16
  ln_to_bf16<D_TOK><<<NT / 8, 256, 0, stream>>>(s, lnqg, lnqb, sn, NT);
  ln_to_bf16<D_ATOM><<<NA / 8, 256, 0, stream>>>(a, lnkg, lnkb, an, NA);

  // 3) K, V   4) Q, G
  kv_gemm<<<NA / 128, 256, 0, stream>>>(an, wkt, wvt, Kb, Vb);
  qg_gemm<<<NT / 64, 256, 0, stream>>>(sn, wqt, wgt, Qf, Gf);

  // 5) ragged attention + gate
  attn_kernel<<<NT / 8, 256, 0, stream>>>(Qf, Gf, Kb, Vb, starts, counts, tmask, attg);

  // 6) output projection
  out_gemm<<<dim3(NT / 64, D_TOK / 128), 256, 0, stream>>>(attg, wot, out);
}
